// EquivariantInteractionBlock_77627238908542
// MI455X (gfx1250) — matrix operand order repaired and checked by reading, not compiled, not run on hardware
//
#include <hip/hip_runtime.h>
#include <hip/hip_fp16.h>
#include <math.h>

#define N_NODES 50000
#define N_EDGES 400000
#define MUL0 16
#define MUL1 8
#define RBF_N 8
#define HID 64
#define WNUM 576          // 16*16 + 8*16 + 16*8 + 8*8
#define XDIM 40           // MUL0 + 3*MUL1
#define CUTOFF 5.0f
#define EPSV 1e-8f
#define A_PATH 0.20412414523193154f   // 1/sqrt(24)
#define INV_SQRT3 0.5773502691896258f
#define INV_SQRT8 0.35355339059327373f

#define EDGE_WAVES 4
#define EDGE_THREADS (EDGE_WAVES*32)
#define N_TILES (N_EDGES/16)          // 25000
#define N_GROUPS (N_TILES/EDGE_WAVES) // 6250

typedef __attribute__((ext_vector_type(16))) _Float16 v16h;
typedef __attribute__((ext_vector_type(8)))  float    v8f;
typedef int v4i __attribute__((vector_size(16)));

union Frag16 { v16h v; _Float16 h[16]; uint32_t u[8]; };

// Fast activation helpers: single v_exp_f32 + single v_rcp_f32 (no IEEE div fixup)
__device__ __forceinline__ float sigm_f(float x) {
  return __builtin_amdgcn_rcpf(1.0f + __expf(-x));
}
__device__ __forceinline__ float silu_f(float x) {
  return x * __builtin_amdgcn_rcpf(1.0f + __expf(-x));
}

// ---------------------------------------------------------------------------
// Kernel 1: irrep RMS norm of x -> xn ; zero aggregation buffers
// ---------------------------------------------------------------------------
__global__ void k_prep_nodes(const float* __restrict__ x, float* __restrict__ xn,
                             float* __restrict__ agg0, float* __restrict__ agg1,
                             float* __restrict__ normw)
{
  int n = blockIdx.x * blockDim.x + threadIdx.x;
  if (n >= N_NODES) return;
  const float* xr = x + (size_t)n * XDIM;
  float s2 = 0.f;
#pragma unroll
  for (int i = 0; i < MUL0; i++) { float v = xr[i]; s2 += v * v; }
  float v2 = 0.f;
#pragma unroll
  for (int i = 0; i < 3*MUL1; i++) { float v = xr[MUL0 + i]; v2 += v * v; }
  float srms = __builtin_amdgcn_rcpf(sqrtf(s2 * (1.0f / MUL0) + EPSV));
  float vrms = __builtin_amdgcn_rcpf(sqrtf(v2 * (1.0f / MUL1) + EPSV));
  float* o = xn + (size_t)n * XDIM;
#pragma unroll
  for (int i = 0; i < MUL0; i++) o[i] = xr[i] * srms;
#pragma unroll
  for (int i = 0; i < 3*MUL1; i++) o[MUL0 + i] = xr[MUL0 + i] * vrms;
#pragma unroll
  for (int i = 0; i < MUL0; i++) agg0[(size_t)n * MUL0 + i] = 0.f;
#pragma unroll
  for (int i = 0; i < 3*MUL1; i++) agg1[(size_t)n * (3*MUL1) + i] = 0.f;
  normw[n] = 0.f;
}

// ---------------------------------------------------------------------------
// Kernel 2: convert MLP weights to f16, transposed [col][k] for B-fragments
// ---------------------------------------------------------------------------
__global__ void k_prep_weights(const float* __restrict__ w_r1,
                               const float* __restrict__ w_r2,
                               const float* __restrict__ w_g1,
                               _Float16* __restrict__ W1T,
                               _Float16* __restrict__ Wg1T,
                               _Float16* __restrict__ W2T)
{
  int idx = blockIdx.x * blockDim.x + threadIdx.x;
  if (idx < HID * 32) {
    int nn = idx >> 5, k = idx & 31;
    W1T[idx]  = (_Float16)((k < RBF_N) ? w_r1[k * HID + nn] : 0.0f);
    Wg1T[idx] = (_Float16)((k < RBF_N) ? w_g1[k * HID + nn] : 0.0f);
  }
  if (idx < WNUM * HID) {
    int col = idx / HID, k = idx % HID;
    W2T[idx] = (_Float16)w_r2[k * WNUM + col];
  }
}

// ---------------------------------------------------------------------------
// Kernel 3: edge kernel. One wave32 per 16-edge tile.
// ---------------------------------------------------------------------------
__global__ __launch_bounds__(EDGE_THREADS) void k_edges(
    const float* __restrict__ xn,
    const int*   __restrict__ esrc, const int* __restrict__ edst,
    const float* __restrict__ esh,  const float* __restrict__ erbf,
    const float* __restrict__ elen,
    const float* __restrict__ b_r1, const float* __restrict__ b_r2,
    const float* __restrict__ b_g1, const float* __restrict__ w_g2,
    const float* __restrict__ b_g2,
    const _Float16* __restrict__ W1T, const _Float16* __restrict__ Wg1T,
    const _Float16* __restrict__ W2T,
    float* __restrict__ agg0, float* __restrict__ agg1, float* __restrict__ normw)
{
  __shared__ _Float16 sW2[WNUM * HID];     // 73728 B
  __shared__ _Float16 sW1[HID * 32];       //  4096 B
  __shared__ _Float16 sWg1[HID * 32];      //  4096 B
  __shared__ float sB2[WNUM];
  __shared__ float sB1[HID], sBg1[HID], sWg2v[HID];
  __shared__ _Float16 sH[EDGE_WAVES][16 * HID];
  __shared__ float sX[EDGE_WAVES][16 * XDIM];     // raw gathered features
  __shared__ float sS0[EDGE_WAVES][16 * MUL0];    // sh0[e] * s_e[i]
  __shared__ float sD2[EDGE_WAVES][16 * MUL1];    // (1/sqrt3) * (v_e[i] . sh1[e])
  __shared__ float sV0[EDGE_WAVES][16 * 3*MUL1];  // sh0[e] * v_e[i][c]
  __shared__ float sSHm[EDGE_WAVES][16 * 4];
  __shared__ int   sSrc[EDGE_WAVES][16];
  __shared__ int   sDst[EDGE_WAVES][16];
  __shared__ float sLen[EDGE_WAVES][16];

  const int tid = threadIdx.x;
  // ---- stage shared weights once per block ----
#if defined(__gfx1250__) && __has_builtin(__builtin_amdgcn_global_load_async_to_lds_b128)
  {
    // CDNA5 async Global->LDS DMA path (tracked by ASYNCcnt)
    typedef __attribute__((address_space(1))) v4i gv4i;
    typedef __attribute__((address_space(3))) v4i lv4i;
    gv4i* g2 = (gv4i*)W2T;  lv4i* l2 = (lv4i*)sW2;
    for (int i = tid; i < (WNUM * HID * 2) / 16; i += EDGE_THREADS)
      __builtin_amdgcn_global_load_async_to_lds_b128(g2 + i, l2 + i, 0, 0);
    gv4i* g1 = (gv4i*)W1T;  lv4i* l1 = (lv4i*)sW1;
    gv4i* gg = (gv4i*)Wg1T; lv4i* lg = (lv4i*)sWg1;
    for (int i = tid; i < (HID * 32 * 2) / 16; i += EDGE_THREADS) {
      __builtin_amdgcn_global_load_async_to_lds_b128(g1 + i, l1 + i, 0, 0);
      __builtin_amdgcn_global_load_async_to_lds_b128(gg + i, lg + i, 0, 0);
    }
    asm volatile("s_wait_asynccnt 0" ::: "memory");
  }
#else
  {
    const uint32_t* s2 = (const uint32_t*)W2T;  uint32_t* d2 = (uint32_t*)sW2;
    for (int i = tid; i < (WNUM * HID) / 2; i += EDGE_THREADS) d2[i] = s2[i];
    const uint32_t* s1 = (const uint32_t*)W1T;  uint32_t* d1 = (uint32_t*)sW1;
    const uint32_t* sg = (const uint32_t*)Wg1T; uint32_t* dg = (uint32_t*)sWg1;
    for (int i = tid; i < (HID * 32) / 2; i += EDGE_THREADS) { d1[i] = s1[i]; dg[i] = sg[i]; }
  }
#endif
  {
    for (int i = tid; i < WNUM; i += EDGE_THREADS) sB2[i] = b_r2[i];
    for (int i = tid; i < HID; i += EDGE_THREADS) { sB1[i] = b_r1[i]; sBg1[i] = b_g1[i]; sWg2v[i] = w_g2[i]; }
  }
  __syncthreads();

  const int wv   = tid >> 5;
  const int lane = tid & 31;
  const int nL   = lane & 15;       // output column within 16-tile / edge idx for staging
  const int hi   = lane >> 4;       // wave half
  const int koff = hi ? 16 : 0;     // B-fragment K offset for this half
  const float bg2 = b_g2[0];
  const int p  = nL >> 3;           // parity for the 8-wide weight blocks
  const int jv = nL & 7;            // m1 output index handled by this lane

  for (int g = blockIdx.x; g < N_GROUPS; g += gridDim.x) {
    const int tile = g * EDGE_WAVES + wv;
    const int e0 = tile * 16;

    // ---- prefetch the next group's edge data into cache (global_prefetch_b8)
    {
      int gn = g + gridDim.x;
      if (gn < N_GROUPS) {
        int en = (gn * EDGE_WAVES + wv) * 16;
        __builtin_prefetch(&erbf[(size_t)en * RBF_N + nL * 16], 0, 0);
        __builtin_prefetch(&esh[(size_t)en * 4 + lane * 2], 0, 0);
        if (lane == 0) {
          __builtin_prefetch(&esrc[en], 0, 0);
          __builtin_prefetch(&edst[en], 0, 0);
          __builtin_prefetch(&elen[en], 0, 0);
        }
      }
    }

    // ---- stage per-tile edge metadata ----
    if (hi == 0) { sSrc[wv][nL] = esrc[e0 + nL]; sLen[wv][nL] = elen[e0 + nL]; }
    else         { sDst[wv][nL] = edst[e0 + nL]; }
    sSHm[wv][lane]      = esh[(size_t)e0 * 4 + lane];
    sSHm[wv][lane + 32] = esh[(size_t)e0 * 4 + lane + 32];
    asm volatile("s_wait_dscnt 0" ::: "memory");

    // ---- gather normalized source-node features ----
    for (int i = lane; i < 16 * XDIM; i += 32) {
      int e = i / XDIM, f = i % XDIM;
      sX[wv][i] = xn[(size_t)sSrc[wv][e] * XDIM + f];
    }
    asm volatile("s_wait_dscnt 0" ::: "memory");

    // ---- coefficient tiles (hoisted out of the WMMA loop) ----
    // sS0[e][i] = sh0[e]*s_e[i]
    for (int i = lane; i < 16 * MUL0; i += 32) {
      int e = i >> 4, ii = i & 15;
      sS0[wv][i] = sSHm[wv][e * 4] * sX[wv][e * XDIM + ii];
    }
    // sD2[e][i] = (1/sqrt3) * v_e[i].sh1[e]
    for (int i = lane; i < 16 * MUL1; i += 32) {
      int e = i >> 3, ii = i & 7;
      const float* xe  = &sX[wv][e * XDIM + MUL0 + ii * 3];
      const float* she = &sSHm[wv][e * 4];
      sD2[wv][i] = INV_SQRT3 * (xe[0] * she[1] + xe[1] * she[2] + xe[2] * she[3]);
    }
    // sV0[e][k] = sh0[e]*v_e flat (k = i*3+c)
    for (int i = lane; i < 16 * 3*MUL1; i += 32) {
      int e = i / (3*MUL1), k = i % (3*MUL1);
      sV0[wv][i] = sSHm[wv][e * 4] * sX[wv][e * XDIM + MUL0 + k];
    }
    asm volatile("s_wait_dscnt 0" ::: "memory");

    // ---- A-fragment: rbf (16x32, K>=8 zero-padded) ----
    Frag16 aR;
#pragma unroll
    for (int j = 0; j < 8; j++) aR.u[j] = 0u;
    if (hi == 0) {
      const float* rb = &erbf[(size_t)(e0 + nL) * RBF_N];
#pragma unroll
      for (int k = 0; k < RBF_N; k++) aR.h[k] = (_Float16)rb[k];
    }

    // ---- hidden layer + gate layer (WMMA, zero C, bias at consumption) ----
    float gpart[8];
#pragma unroll
    for (int r = 0; r < 8; r++) gpart[r] = 0.f;
#pragma unroll
    for (int nt = 0; nt < HID / 16; nt++) {
      Frag16 bf, bgf;
      const _Float16* c1 = &sW1[(nt * 16 + nL) * 32];
      const _Float16* cg = &sWg1[(nt * 16 + nL) * 32];
#pragma unroll
      for (int j = 0; j < 8; j++) {
        bf.u[j]  = *(const uint32_t*)&c1[koff + 2 * j];
        bgf.u[j] = *(const uint32_t*)&cg[koff + 2 * j];
      }
      v8f zc = {};
      v8f hc = __builtin_amdgcn_wmma_f32_16x16x32_f16(false, aR.v, false, bf.v,  (short)0, zc, false, false);
      v8f gc = __builtin_amdgcn_wmma_f32_16x16x32_f16(false, aR.v, false, bgf.v, (short)0, zc, false, false);
      float bb  = sB1[nt * 16 + nL];
      float bbg = sBg1[nt * 16 + nL];
      float wg  = sWg2v[nt * 16 + nL];
#pragma unroll
      for (int r = 0; r < 8; r++) {
        sH[wv][(r + 8 * hi) * HID + nt * 16 + nL] = (_Float16)silu_f(hc[r] + bb);
        gpart[r] += silu_f(gc[r] + bbg) * wg;
      }
    }
    // reduce gate dot-product across the 16 lanes of each half
#pragma unroll
    for (int m = 1; m < 16; m <<= 1) {
#pragma unroll
      for (int r = 0; r < 8; r++) gpart[r] += __shfl_xor(gpart[r], m, 32);
    }
    float ew[8];
#pragma unroll
    for (int r = 0; r < 8; r++) {
      float len = sLen[wv][r + 8 * hi];
      float cw = (len < CUTOFF) ? 0.5f * (__cosf(3.14159265358979f * len * (1.0f / CUTOFF)) + 1.0f) : 0.0f;
      ew[r] = cw * sigm_f(gpart[r] + bg2);
    }
    asm volatile("s_wait_dscnt 0" ::: "memory");

    // ---- A-fragments of h (K = 0..31 and 32..63) ----
    Frag16 a0, a1;
    {
      const _Float16* hrow = &sH[wv][nL * HID];
      const int khalf = hi ? 8 : 0;
#pragma unroll
      for (int j = 0; j < 8; j++) {
        int kidx = koff + 2 * j;
        a0.u[j] = *(const uint32_t*)&hrow[kidx];
        a1.u[j] = *(const uint32_t*)&hrow[32 + kidx];
      }
    }

    // ---- second GEMM: 36 col-tiles x 2 K-steps, messages from D-fragments ----
    float m0acc[8], m1acc[8][3];
#pragma unroll
    for (int r = 0; r < 8; r++) { m0acc[r] = 0.f; m1acc[r][0] = 0.f; m1acc[r][1] = 0.f; m1acc[r][2] = 0.f; }

#define WTILE(nt2_, cvar_)                                                            \
    v8f cvar_;                                                                        \
    {                                                                                 \
      Frag16 b0f, b1f;                                                                \
      const _Float16* wc = &sW2[((nt2_) * 16 + nL) * HID];                            \
      _Pragma("unroll")                                                               \
      for (int j = 0; j < 8; j++) {                                                   \
        b0f.u[j] = *(const uint32_t*)&wc[koff + 2 * j];                               \
        b1f.u[j] = *(const uint32_t*)&wc[32 + koff + 2 * j];                          \
      }                                                                               \
      v8f zc_ = {};                                                                   \
      cvar_ = __builtin_amdgcn_wmma_f32_16x16x32_f16(false, a0.v, false, b0f.v, (short)0, zc_, false, false); \
      cvar_ = __builtin_amdgcn_wmma_f32_16x16x32_f16(false, a1.v, false, b1f.v, (short)0, cvar_, false, false); \
    }

    // w1 block: cols [0,256), coef = sS0[e][nt2]
#pragma unroll 2
    for (int nt2 = 0; nt2 < 16; nt2++) {
      WTILE(nt2, c)
      float bb = sB2[nt2 * 16 + nL];
#pragma unroll
      for (int r = 0; r < 8; r++) {
        int e = r + 8 * hi;
        m0acc[r] += sS0[wv][e * MUL0 + nt2] * (c[r] + bb);
      }
    }
    // w2 block: cols [256,384), coef = sD2[e][nt2-16]
#pragma unroll 2
    for (int nt2 = 16; nt2 < 24; nt2++) {
      WTILE(nt2, c)
      float bb = sB2[nt2 * 16 + nL];
      const int i = nt2 - 16;
#pragma unroll
      for (int r = 0; r < 8; r++) {
        int e = r + 8 * hi;
        m0acc[r] += sD2[wv][e * MUL1 + i] * (c[r] + bb);
      }
    }
    // w3 block: cols [384,512), i = 2*(nt2-24)+p, coef = s_e[i], times sh1[c]
#pragma unroll 2
    for (int nt2 = 24; nt2 < 32; nt2++) {
      WTILE(nt2, c)
      float bb = sB2[nt2 * 16 + nL];
      const int i = 2 * (nt2 - 24) + p;
#pragma unroll
      for (int r = 0; r < 8; r++) {
        int e = r + 8 * hi;
        float t = sX[wv][e * XDIM + i] * (c[r] + bb);
        m1acc[r][0] += t * sSHm[wv][e * 4 + 1];
        m1acc[r][1] += t * sSHm[wv][e * 4 + 2];
        m1acc[r][2] += t * sSHm[wv][e * 4 + 3];
      }
    }
    // w4 block: cols [512,576), i = 2*(nt2-32)+p, coef = sh0*v_e[i][c]
#pragma unroll 2
    for (int nt2 = 32; nt2 < 36; nt2++) {
      WTILE(nt2, c)
      float bb = sB2[nt2 * 16 + nL];
      const int i = 2 * (nt2 - 32) + p;
#pragma unroll
      for (int r = 0; r < 8; r++) {
        int e = r + 8 * hi;
        float wcb = c[r] + bb;
        const float* ve = &sV0[wv][e * (3*MUL1) + i * 3];
        m1acc[r][0] += wcb * ve[0];
        m1acc[r][1] += wcb * ve[1];
        m1acc[r][2] += wcb * ve[2];
      }
    }
#undef WTILE

    // combine the two parity lanes' m1 partials (halves the atomic traffic)
#pragma unroll
    for (int r = 0; r < 8; r++) {
      m1acc[r][0] += __shfl_xor(m1acc[r][0], 8, 32);
      m1acc[r][1] += __shfl_xor(m1acc[r][1], 8, 32);
      m1acc[r][2] += __shfl_xor(m1acc[r][2], 8, 32);
    }

    // ---- scatter with float atomics (global_atomic_add_f32) ----
#pragma unroll
    for (int r = 0; r < 8; r++) {
      int e = r + 8 * hi;
      int dst = sDst[wv][e];
      float scl = A_PATH * ew[r];
      unsafeAtomicAdd(&agg0[(size_t)dst * MUL0 + nL], m0acc[r] * scl);
      if (p == 0) {
        unsafeAtomicAdd(&agg1[(size_t)dst * (3*MUL1) + jv * 3 + 0], m1acc[r][0] * scl);
        unsafeAtomicAdd(&agg1[(size_t)dst * (3*MUL1) + jv * 3 + 1], m1acc[r][1] * scl);
        unsafeAtomicAdd(&agg1[(size_t)dst * (3*MUL1) + jv * 3 + 2], m1acc[r][2] * scl);
      }
      if (nL == 0) unsafeAtomicAdd(&normw[dst], ew[r]);
    }
  }
}

// ---------------------------------------------------------------------------
// Kernel 4: per-node output MLP + gating + residual
// ---------------------------------------------------------------------------
__global__ void k_nodes(const float* __restrict__ x, const float* __restrict__ xnrm,
                        const float* __restrict__ agg0, const float* __restrict__ agg1,
                        const float* __restrict__ normw,
                        const float* __restrict__ Wm_s, const float* __restrict__ Wm_v,
                        const float* __restrict__ Wu_s, const float* __restrict__ Wu_v,
                        const float* __restrict__ Ws_s, const float* __restrict__ Ws_v,
                        const float* __restrict__ res_scale,
                        float* __restrict__ out)
{
  __shared__ float sWm_s[MUL0 * (MUL0 + MUL1)];
  __shared__ float sWm_v[MUL1 * MUL1];
  __shared__ float sWu_s[MUL0 * MUL0];
  __shared__ float sWu_v[MUL1 * MUL1];
  __shared__ float sWs_s[MUL0 * MUL0];
  __shared__ float sWs_v[MUL1 * MUL1];
  int tid = threadIdx.x;
  for (int i = tid; i < MUL0 * (MUL0 + MUL1); i += blockDim.x) sWm_s[i] = Wm_s[i];
  for (int i = tid; i < MUL1 * MUL1; i += blockDim.x) { sWm_v[i] = Wm_v[i]; sWu_v[i] = Wu_v[i]; sWs_v[i] = Ws_v[i]; }
  for (int i = tid; i < MUL0 * MUL0; i += blockDim.x) { sWu_s[i] = Wu_s[i]; sWs_s[i] = Ws_s[i]; }
  __syncthreads();

  int n = blockIdx.x * blockDim.x + tid;
  if (n >= N_NODES) return;
  float nv = normw[n]; nv = (nv > EPSV) ? nv : EPSV;
  float inv = __builtin_amdgcn_rcpf(nv);
  float a0[MUL0];
#pragma unroll
  for (int i = 0; i < MUL0; i++) a0[i] = agg0[(size_t)n * MUL0 + i] * inv;
  float a1[MUL1][3];
#pragma unroll
  for (int i = 0; i < MUL1; i++) {
    a1[i][0] = agg1[(size_t)n * 24 + i * 3 + 0] * inv;
    a1[i][1] = agg1[(size_t)n * 24 + i * 3 + 1] * inv;
    a1[i][2] = agg1[(size_t)n * 24 + i * 3 + 2] * inv;
  }
  float scal[MUL0], gates[MUL1];
#pragma unroll
  for (int j = 0; j < MUL0 + MUL1; j++) {
    float s = 0.f;
#pragma unroll
    for (int i = 0; i < MUL0; i++) s += a0[i] * sWm_s[i * (MUL0 + MUL1) + j];
    s *= 0.25f;                             // 1/sqrt(MUL0)
    if (j < MUL0) scal[j] = silu_f(s); else gates[j - MUL0] = sigm_f(s);
  }
  float vg[MUL1][3];
#pragma unroll
  for (int j = 0; j < MUL1; j++) {
    float v0 = 0.f, v1 = 0.f, v2 = 0.f;
#pragma unroll
    for (int i = 0; i < MUL1; i++) {
      float w = sWm_v[i * MUL1 + j];
      v0 += a1[i][0] * w; v1 += a1[i][1] * w; v2 += a1[i][2] * w;
    }
    float gt = gates[j] * INV_SQRT8;        // fold 1/sqrt(MUL1) of Wm_v
    vg[j][0] = v0 * gt; vg[j][1] = v1 * gt; vg[j][2] = v2 * gt;
  }
  const float* xr = x + (size_t)n * XDIM;
  const float* xs = xnrm + (size_t)n * XDIM;
  float rs = res_scale[0];
#pragma unroll
  for (int j = 0; j < MUL0; j++) {
    float s = 0.f;
#pragma unroll
    for (int i = 0; i < MUL0; i++) s += scal[i] * sWu_s[i * MUL0 + j] + xs[i] * sWs_s[i * MUL0 + j];
    out[(size_t)n * XDIM + j] = xr[j] + rs * 0.25f * s;
  }
#pragma unroll
  for (int j = 0; j < MUL1; j++) {
    float o0 = 0.f, o1 = 0.f, o2 = 0.f;
#pragma unroll
    for (int i = 0; i < MUL1; i++) {
      float wu = sWu_v[i * MUL1 + j], ws = sWs_v[i * MUL1 + j];
      o0 += vg[i][0] * wu + xs[MUL0 + i * 3 + 0] * ws;
      o1 += vg[i][1] * wu + xs[MUL0 + i * 3 + 1] * ws;
      o2 += vg[i][2] * wu + xs[MUL0 + i * 3 + 2] * ws;
    }
    out[(size_t)n * XDIM + MUL0 + j * 3 + 0] = xr[MUL0 + j * 3 + 0] + rs * INV_SQRT8 * o0;
    out[(size_t)n * XDIM + MUL0 + j * 3 + 1] = xr[MUL0 + j * 3 + 1] + rs * INV_SQRT8 * o1;
    out[(size_t)n * XDIM + MUL0 + j * 3 + 2] = xr[MUL0 + j * 3 + 2] + rs * INV_SQRT8 * o2;
  }
}

// ---------------------------------------------------------------------------
extern "C" void kernel_launch(void* const* d_in, const int* in_sizes, int n_in,
                              void* d_out, int out_size, void* d_ws, size_t ws_size,
                              hipStream_t stream)
{
  const float* x    = (const float*)d_in[0];
  const int*   esrc = (const int*)d_in[1];
  const int*   edst = (const int*)d_in[2];
  const float* esh  = (const float*)d_in[3];
  const float* erbf = (const float*)d_in[4];
  const float* elen = (const float*)d_in[5];
  const float* w_r1 = (const float*)d_in[6];
  const float* b_r1 = (const float*)d_in[7];
  const float* w_r2 = (const float*)d_in[8];
  const float* b_r2 = (const float*)d_in[9];
  const float* w_g1 = (const float*)d_in[10];
  const float* b_g1 = (const float*)d_in[11];
  const float* w_g2 = (const float*)d_in[12];
  const float* b_g2 = (const float*)d_in[13];
  const float* Wm_s = (const float*)d_in[14];
  const float* Wm_v = (const float*)d_in[15];
  const float* Wu_s = (const float*)d_in[16];
  const float* Wu_v = (const float*)d_in[17];
  const float* Ws_s = (const float*)d_in[18];
  const float* Ws_v = (const float*)d_in[19];
  const float* rsc  = (const float*)d_in[20];
  float* out = (float*)d_out;
  (void)in_sizes; (void)n_in; (void)out_size; (void)ws_size;

  char* ws = (char*)d_ws;
  size_t off = 0;
  auto wsalloc = [&](size_t bytes) -> void* {
    void* p = ws + off;
    off = (off + bytes + 255) & ~(size_t)255;
    return p;
  };
  float*    xnrm  = (float*)wsalloc((size_t)N_NODES * XDIM * 4);
  float*    agg0  = (float*)wsalloc((size_t)N_NODES * MUL0 * 4);
  float*    agg1  = (float*)wsalloc((size_t)N_NODES * 3 * MUL1 * 4);
  float*    normw = (float*)wsalloc((size_t)N_NODES * 4);
  _Float16* W1T   = (_Float16*)wsalloc((size_t)HID * 32 * 2);
  _Float16* Wg1T  = (_Float16*)wsalloc((size_t)HID * 32 * 2);
  _Float16* W2T   = (_Float16*)wsalloc((size_t)WNUM * HID * 2);

  hipLaunchKernelGGL(k_prep_nodes, dim3((N_NODES + 255) / 256), dim3(256), 0, stream,
                     x, xnrm, agg0, agg1, normw);
  hipLaunchKernelGGL(k_prep_weights, dim3((WNUM * HID + 255) / 256), dim3(256), 0, stream,
                     w_r1, w_r2, w_g1, W1T, Wg1T, W2T);
  hipLaunchKernelGGL(k_edges, dim3(1024), dim3(EDGE_THREADS), 0, stream,
                     xnrm, esrc, edst, esh, erbf, elen,
                     b_r1, b_r2, b_g1, w_g2, b_g2,
                     W1T, Wg1T, W2T, agg0, agg1, normw);
  hipLaunchKernelGGL(k_nodes, dim3((N_NODES + 255) / 256), dim3(256), 0, stream,
                     x, xnrm, agg0, agg1, normw,
                     Wm_s, Wm_v, Wu_s, Wu_v, Ws_s, Ws_v, rsc, out);
}
